// MLPMixerLayer_37838661878375
// MI455X (gfx1250) — matrix operand order repaired and checked by reading, not compiled, not run on hardware
//
#include <hip/hip_runtime.h>
#include <hip/hip_bf16.h>

#define BDIM 256
#define PDIM 256
#define CDIM 512
#define LN_EPS 1e-6f

typedef _Float16 half_t;
typedef __attribute__((ext_vector_type(8)))  _Float16 v8h;
typedef __attribute__((ext_vector_type(16))) _Float16 v16h;
typedef __attribute__((ext_vector_type(8)))  float    v8f;

__device__ __forceinline__ v16h cat16(v8h lo, v8h hi) {
  union { v16h v; v8h h[2]; } u;
  u.h[0] = lo; u.h[1] = hi;
  return u.v;
}

__device__ __forceinline__ float gelu_exact(float v) {
  return 0.5f * v * (1.0f + erff(v * 0.70710678118654752f));
}

#define WMMA_F16(A, B, C) \
  __builtin_amdgcn_wmma_f32_16x16x32_f16(false, (A), false, (B), (short)0, (C), false, false)

// ---------------------------------------------------------------------------
// Transpose + fp32 -> f16 convert for weight matrices: dst[c][r] = src[r][c]
// ---------------------------------------------------------------------------
__global__ void transpose_cvt(const float* __restrict__ src, half_t* __restrict__ dst,
                              int rows, int cols) {
  int idx = blockIdx.x * blockDim.x + threadIdx.x;
  if (idx >= rows * cols) return;
  int rr = idx / cols, cc = idx - rr * cols;
  dst[cc * rows + rr] = (half_t)src[rr * cols + cc];
}

// ---------------------------------------------------------------------------
// LayerNorm over last dim (C=512). One wave per row; 16 elements per lane.
// ---------------------------------------------------------------------------
__global__ __launch_bounds__(256)
void ln_kernel(const float* __restrict__ src,
               const float* __restrict__ g,
               const float* __restrict__ beta,
               half_t* __restrict__ out16,
               float*  __restrict__ out32) {
  const int lane = threadIdx.x & 31;
  const int row  = (blockIdx.x * blockDim.x + threadIdx.x) >> 5;  // wave-uniform
  const float* p = src + (size_t)row * CDIM;

  float v[16];
  float s = 0.f;
#pragma unroll
  for (int j = 0; j < 16; ++j) { v[j] = p[j * 32 + lane]; s += v[j]; }
#pragma unroll
  for (int m = 16; m >= 1; m >>= 1) s += __shfl_xor(s, m, 32);
  const float mu = s * (1.0f / CDIM);

  float q = 0.f;
#pragma unroll
  for (int j = 0; j < 16; ++j) { float d = v[j] - mu; q += d * d; }
#pragma unroll
  for (int m = 16; m >= 1; m >>= 1) q += __shfl_xor(q, m, 32);
  const float rs = rsqrtf(q * (1.0f / CDIM) + LN_EPS);

#pragma unroll
  for (int j = 0; j < 16; ++j) {
    const int c = j * 32 + lane;
    const float h = (v[j] - mu) * rs * g[c] + beta[c];
    out16[(size_t)row * CDIM + c] = (half_t)h;
    if (out32) out32[(size_t)row * CDIM + c] = h;
  }
}

// ---------------------------------------------------------------------------
// Fused token-mixing MLP. One block = one batch x one 64-column slab of C.
//   stage 1:  G  = gelu(w1a^T @ h + b1a)          [256 x 64]  -> LDS gT[n][k]
//   stage 2:  x2 = w1b^T @ G + b1b + x            [256 x 64]  -> global fp32
// The GELU intermediate never touches global memory.
// 8 waves; wave tile 64(M) x 32(N) per stage (8 WMMAs per K-step).
// ---------------------------------------------------------------------------
__global__ __launch_bounds__(256)
void token_fused(const half_t* __restrict__ W1at,  // [P,P] f16, pre-transposed
                 const half_t* __restrict__ W1bt,  // [P,P] f16, pre-transposed
                 const half_t* __restrict__ Act,   // h1 f16 [B,P,C]
                 const float*  __restrict__ b1a,
                 const float*  __restrict__ b1b,
                 const float*  __restrict__ x,     // residual fp32 [B,P,C]
                 float* __restrict__ x2out) {
  __shared__ half_t hT[64 * 40];    // per-K-chunk activation stage, [n][k32]
  __shared__ half_t gT[64 * 264];   // stage-1 output, [n][m=256] (+pad)

  const int batch  = blockIdx.z;
  const int blockN = blockIdx.x * 64;
  const int tid  = threadIdx.x;
  const int lane = tid & 31, wave = tid >> 5;
  const int r = lane & 15, hi = lane >> 4;
  const int wm = wave & 3, wn = wave >> 2;
  const int n0 = wn * 32;

  const half_t* actb = Act + (size_t)batch * PDIM * CDIM;
  const int krow = tid >> 3;        // 0..31
  const int cofs = (tid & 7) * 8;   // 0..56

  // -------- stage 1 --------
  {
    v8f acc[4][2] = {};
    for (int kk = 0; kk < PDIM; kk += 32) {
      // stage 32(K) x 64(N) activation chunk, transposed into hT[n][k]
      v8h ld = *(const v8h*)(actb + (size_t)(kk + krow) * CDIM + blockN + cofs);
#pragma unroll
      for (int j = 0; j < 8; ++j) hT[(cofs + j) * 40 + krow] = ld[j];
      __syncthreads();

      v16h a[4], b[2];
#pragma unroll
      for (int mi = 0; mi < 4; ++mi) {
        const int m0 = (mi >> 1) * 128 + wm * 32 + (mi & 1) * 16;
        const half_t* ap = W1at + (size_t)(m0 + r) * PDIM + kk;
        a[mi] = cat16(*(const v8h*)(ap + hi * 16), *(const v8h*)(ap + hi * 16 + 8));
      }
#pragma unroll
      for (int nt = 0; nt < 2; ++nt) {
        const half_t* bp = &hT[(n0 + nt * 16 + r) * 40 + hi * 16];
        b[nt] = cat16(*(const v8h*)bp, *(const v8h*)(bp + 8));
      }
#pragma unroll
      for (int mi = 0; mi < 4; ++mi)
#pragma unroll
        for (int nt = 0; nt < 2; ++nt)
          acc[mi][nt] = WMMA_F16(a[mi], b[nt], acc[mi][nt]);
      __syncthreads();
    }
    // epilogue: bias + gelu, store transposed into gT[n][m]
#pragma unroll
    for (int mi = 0; mi < 4; ++mi) {
      const int m0 = (mi >> 1) * 128 + wm * 32 + (mi & 1) * 16;
#pragma unroll
      for (int nt = 0; nt < 2; ++nt) {
        const int n = n0 + nt * 16 + r;
#pragma unroll
        for (int i = 0; i < 8; ++i) {
          const int m = m0 + i + 8 * hi;
          gT[n * 264 + m] = (half_t)gelu_exact(acc[mi][nt][i] + b1a[m]);
        }
      }
    }
  }
  __syncthreads();

  // -------- stage 2 --------
  {
    const float* xb = x     + (size_t)batch * PDIM * CDIM;
    float*       ob = x2out + (size_t)batch * PDIM * CDIM;
    v8f acc[4][2] = {};
    for (int kk = 0; kk < PDIM; kk += 32) {
      v16h a[4], b[2];
#pragma unroll
      for (int mi = 0; mi < 4; ++mi) {
        const int m0 = (mi >> 1) * 128 + wm * 32 + (mi & 1) * 16;
        const half_t* ap = W1bt + (size_t)(m0 + r) * PDIM + kk;
        a[mi] = cat16(*(const v8h*)(ap + hi * 16), *(const v8h*)(ap + hi * 16 + 8));
      }
#pragma unroll
      for (int nt = 0; nt < 2; ++nt) {
        const half_t* bp = &gT[(n0 + nt * 16 + r) * 264 + kk + hi * 16];
        b[nt] = cat16(*(const v8h*)bp, *(const v8h*)(bp + 8));
      }
#pragma unroll
      for (int mi = 0; mi < 4; ++mi)
#pragma unroll
        for (int nt = 0; nt < 2; ++nt)
          acc[mi][nt] = WMMA_F16(a[mi], b[nt], acc[mi][nt]);
    }
    // epilogue: bias + residual -> x2 (fp32)
#pragma unroll
    for (int mi = 0; mi < 4; ++mi) {
      const int m0 = (mi >> 1) * 128 + wm * 32 + (mi & 1) * 16;
#pragma unroll
      for (int nt = 0; nt < 2; ++nt) {
        const int n = blockN + n0 + nt * 16 + r;
#pragma unroll
        for (int i = 0; i < 8; ++i) {
          const int m = m0 + i + 8 * hi;
          const size_t idx = (size_t)m * CDIM + n;
          ob[idx] = acc[mi][nt][i] + b1b[m] + xb[idx];
        }
      }
    }
  }
}

// ---------------------------------------------------------------------------
// Fused channel-mixing MLP. One block = one batch x one 32-row slab of P,
// full C=512 width. GELU intermediate y2 lives entirely in LDS row-major
// (A-operand friendly for stage 2).
//   stage 1:  y2  = gelu(h2 @ w2a + b2a)     [32 x 512] -> LDS
//   stage 2:  out = h2_f32 + y2 @ w2b + b2b  [32 x 512] -> d_out
// 8 waves; wave tile 32(M) x 64(N); all waves share the same A rows.
// ---------------------------------------------------------------------------
__global__ __launch_bounds__(256)
void chan_fused(const half_t* __restrict__ Act,   // h2 f16 [B,P,C]
                const half_t* __restrict__ W2at,  // [C,C] f16, pre-transposed
                const half_t* __restrict__ W2bt,  // [C,C] f16, pre-transposed
                const float*  __restrict__ b2a,
                const float*  __restrict__ b2b,
                float* __restrict__ inout) {      // d_out: holds h2 f32; += t2
  __shared__ half_t y2[32 * 520];   // [m][k=512] (+pad)

  const int batch  = blockIdx.z;
  const int blockM = blockIdx.y * 32;
  const int lane = threadIdx.x & 31, wave = threadIdx.x >> 5;
  const int r = lane & 15, hi = lane >> 4;
  const int n0 = wave * 64;         // each wave owns a 64-wide N strip

  const half_t* ab = Act + (size_t)batch * PDIM * CDIM;

  // -------- stage 1 --------
  {
    v8f acc[2][4] = {};
    for (int kk = 0; kk < CDIM; kk += 32) {
      v16h a[2], b[4];
#pragma unroll
      for (int mt = 0; mt < 2; ++mt) {
        const half_t* ap = ab + (size_t)(blockM + mt * 16 + r) * CDIM + kk;
        a[mt] = cat16(*(const v8h*)(ap + hi * 16), *(const v8h*)(ap + hi * 16 + 8));
      }
#pragma unroll
      for (int nt = 0; nt < 4; ++nt) {
        const half_t* bp = W2at + (size_t)(n0 + nt * 16 + r) * CDIM + kk + hi * 16;
        b[nt] = cat16(*(const v8h*)bp, *(const v8h*)(bp + 8));
      }
#pragma unroll
      for (int mt = 0; mt < 2; ++mt)
#pragma unroll
        for (int nt = 0; nt < 4; ++nt)
          acc[mt][nt] = WMMA_F16(a[mt], b[nt], acc[mt][nt]);
    }
    // epilogue: bias + gelu -> LDS y2[m][n] (row-major over K of stage 2)
#pragma unroll
    for (int mt = 0; mt < 2; ++mt)
#pragma unroll
      for (int nt = 0; nt < 4; ++nt) {
        const int n = n0 + nt * 16 + r;
        const float bn = b2a[n];
#pragma unroll
        for (int i = 0; i < 8; ++i) {
          const int m = mt * 16 + i + 8 * hi;
          y2[m * 520 + n] = (half_t)gelu_exact(acc[mt][nt][i] + bn);
        }
      }
  }
  __syncthreads();

  // -------- stage 2 --------
  {
    float* ob = inout + (size_t)batch * PDIM * CDIM;
    v8f acc[2][4] = {};
    for (int kk = 0; kk < CDIM; kk += 32) {
      v16h a[2], b[4];
#pragma unroll
      for (int mt = 0; mt < 2; ++mt) {
        const half_t* ap = &y2[(mt * 16 + r) * 520 + kk];
        a[mt] = cat16(*(const v8h*)(ap + hi * 16), *(const v8h*)(ap + hi * 16 + 8));
      }
#pragma unroll
      for (int nt = 0; nt < 4; ++nt) {
        const half_t* bp = W2bt + (size_t)(n0 + nt * 16 + r) * CDIM + kk + hi * 16;
        b[nt] = cat16(*(const v8h*)bp, *(const v8h*)(bp + 8));
      }
#pragma unroll
      for (int mt = 0; mt < 2; ++mt)
#pragma unroll
        for (int nt = 0; nt < 4; ++nt)
          acc[mt][nt] = WMMA_F16(a[mt], b[nt], acc[mt][nt]);
    }
    // epilogue: out = h2 + t2 + b2b
#pragma unroll
    for (int mt = 0; mt < 2; ++mt)
#pragma unroll
      for (int nt = 0; nt < 4; ++nt) {
        const int n = n0 + nt * 16 + r;
        const float bn = b2b[n];
#pragma unroll
        for (int i = 0; i < 8; ++i) {
          const int m = blockM + mt * 16 + i + 8 * hi;
          const size_t idx = (size_t)m * CDIM + n;
          ob[idx] = ob[idx] + acc[mt][nt][i] + bn;
        }
      }
  }
}

// ---------------------------------------------------------------------------
extern "C" void kernel_launch(void* const* d_in, const int* in_sizes, int n_in,
                              void* d_out, int out_size, void* d_ws, size_t ws_size,
                              hipStream_t stream) {
  const float* x    = (const float*)d_in[0];
  const float* ln_g = (const float*)d_in[1];
  const float* ln_b = (const float*)d_in[2];
  const float* w1a  = (const float*)d_in[3];
  const float* b1a  = (const float*)d_in[4];
  const float* w1b  = (const float*)d_in[5];
  const float* b1b  = (const float*)d_in[6];
  const float* w2a  = (const float*)d_in[7];
  const float* b2a  = (const float*)d_in[8];
  const float* w2b  = (const float*)d_in[9];
  const float* b2b  = (const float*)d_in[10];
  float* out = (float*)d_out;

  const size_t NE = (size_t)BDIM * PDIM * CDIM;  // 33,554,432 elements
  char* ws = (char*)d_ws;
  half_t* hbuf = (half_t*)ws;              // NE f16: h1, then reused for h2
  float*  x2   = (float*)(ws + NE * 2);    // NE f32: token-mix output
  half_t* w1at = (half_t*)(ws + NE * 6);   // transposed f16 weights
  half_t* w1bt = w1at + PDIM * PDIM;
  half_t* w2at = w1bt + PDIM * PDIM;
  half_t* w2bt = w2at + CDIM * CDIM;

  // 1) weights: transpose + convert to f16 (stay hot in L2 across GEMMs)
  transpose_cvt<<<(PDIM * PDIM) / 256, 256, 0, stream>>>(w1a, w1at, PDIM, PDIM);
  transpose_cvt<<<(PDIM * PDIM) / 256, 256, 0, stream>>>(w1b, w1bt, PDIM, PDIM);
  transpose_cvt<<<(CDIM * CDIM) / 256, 256, 0, stream>>>(w2a, w2at, CDIM, CDIM);
  transpose_cvt<<<(CDIM * CDIM) / 256, 256, 0, stream>>>(w2b, w2bt, CDIM, CDIM);

  // 2) h1 = LN(x) -> f16
  ln_kernel<<<(BDIM * PDIM) / 8, 256, 0, stream>>>(x, ln_g, ln_b, hbuf, nullptr);

  // 3) fused token mixing: x2 = w1b^T @ gelu(w1a^T @ h1 + b1a) + b1b + x
  token_fused<<<dim3(CDIM / 64, 1, BDIM), 256, 0, stream>>>(
      w1at, w1bt, hbuf, b1a, b1b, x, x2);

  // 4) h2 = LN(x2): f16 for GEMM, f32 straight into d_out (residual base)
  ln_kernel<<<(BDIM * PDIM) / 8, 256, 0, stream>>>(x2, ln_g, ln_b, hbuf, out);

  // 5) fused channel mixing: out = h2 + gelu(h2 @ w2a + b2a) @ w2b + b2b
  chan_fused<<<dim3(1, PDIM / 32, BDIM), 256, 0, stream>>>(
      hbuf, w2at, w2bt, b2a, b2b, out);
}
